// DNA_37022618091708
// MI455X (gfx1250) — compile-verified, matrix operand order checked by reading, not run on hardware
//
#include <hip/hip_runtime.h>
#include <math.h>

// ---------------------------------------------------------------------------
// Problem dims (from reference)
// ---------------------------------------------------------------------------
namespace {
constexpr int kT    = 4096;
constexpr int kD    = 1024;
constexpr int kH    = 16;
constexpr int kDH   = 64;
constexpr int kE    = 8;
constexpr int kCAP  = 1024;
constexpr int kMLP  = 4096;
constexpr int kHOPS = 3;
constexpr int kNR   = kE + 1;   // router outputs (9)
}

typedef __attribute__((ext_vector_type(16))) __bf16 v16bf;
typedef __attribute__((ext_vector_type(8)))  float  v8f;

// ---------------------------------------------------------------------------
// CDNA5 async global->LDS copy (ASYNCcnt path), guarded so compile never fails
// ---------------------------------------------------------------------------
#if defined(__has_builtin)
#  if __has_builtin(__builtin_amdgcn_global_load_async_to_lds_b128)
#    define HAVE_ASYNC_COPY 1
#  endif
#  if __has_builtin(__builtin_amdgcn_s_wait_asynccnt)
#    define HAVE_WAIT_ASYNC 1
#  endif
#  if __has_builtin(__builtin_amdgcn_tensor_load_to_lds) && \
      __has_builtin(__builtin_amdgcn_s_wait_tensorcnt)
#    define HAVE_TDM 1
#  endif
#endif

#if defined(HAVE_ASYNC_COPY)
typedef int v4i_vs __attribute__((vector_size(16)));
typedef __attribute__((address_space(1))) v4i_vs* as1_v4i;
typedef __attribute__((address_space(3))) v4i_vs* as3_v4i;
__device__ __forceinline__ void g2lds_b128(const unsigned short* g, unsigned short* l) {
  __builtin_amdgcn_global_load_async_to_lds_b128(
      (as1_v4i)(void*)g, (as3_v4i)(void*)l, 0, 0);
}
#endif

__device__ __forceinline__ void wait_async_n(int n) {
#if defined(HAVE_ASYNC_COPY)
#  if defined(HAVE_WAIT_ASYNC)
  switch (n) {
    case 0:  __builtin_amdgcn_s_wait_asynccnt(0); break;
    case 2:  __builtin_amdgcn_s_wait_asynccnt(2); break;
    default: __builtin_amdgcn_s_wait_asynccnt(4); break;
  }
#  else
  switch (n) {
    case 0:  asm volatile("s_wait_asynccnt 0x0" ::: "memory"); break;
    case 2:  asm volatile("s_wait_asynccnt 0x2" ::: "memory"); break;
    default: asm volatile("s_wait_asynccnt 0x4" ::: "memory"); break;
  }
#  endif
#else
  (void)n;
#endif
}

// ---------------------------------------------------------------------------
// Tensor Data Mover: 1D tile copy global -> LDS (ISA 08_async_tensor §8).
// D# group0: count=1 | lds_addr | global_addr[56:0] | type=2
// D# group1: data_size | tensor_dim0=N | tensor_dim1=1 | tile_dim0=N |
//            tile_dim1=1 | tensor_dim0_stride=N
// ---------------------------------------------------------------------------
#if defined(HAVE_TDM)
typedef unsigned int u32x4 __attribute__((vector_size(16)));
typedef int          i32x4 __attribute__((vector_size(16)));
typedef int          i32x8 __attribute__((vector_size(32)));

__device__ __forceinline__ void tdm_load_1d(const void* gsrc, void* ldsDst,
                                            int nElems, int dataSizeCode) {
  unsigned long long ga = (unsigned long long)(size_t)gsrc;
  unsigned lo = (unsigned)(unsigned long long)
      (__attribute__((address_space(3))) char*)ldsDst;   // raw LDS byte offset
  u32x4 g0;
  g0[0] = 1u;                                            // count=1 (valid D#)
  g0[1] = lo;                                            // lds_addr
  g0[2] = (unsigned)(ga & 0xFFFFFFFFu);                  // global_addr[31:0]
  g0[3] = (unsigned)((ga >> 32) & 0x01FFFFFFu)           // global_addr[56:32]
          | (2u << 30);                                  // type=2 ("image")
  i32x8 g1;
  g1[0] = dataSizeCode << 16;                            // wg_mask=0, data_size
  g1[1] = (nElems & 0xFFFF) << 16;                       // tensor_dim0[15:0] @bit48
  g1[2] = ((nElems >> 16) & 0xFFFF) | (1 << 16);         // tensor_dim0[31:16], dim1=1
  g1[3] = (nElems & 0xFFFF) << 16;                       // tile_dim0 = nElems
  g1[4] = 1;                                             // tile_dim1=1, tile_dim2=0
  g1[5] = nElems;                                        // tensor_dim0_stride[31:0]
  g1[6] = 0;
  g1[7] = 0;
  i32x4 z4 = {0, 0, 0, 0};
#if defined(__clang_major__) && (__clang_major__ >= 23)
  i32x8 z8 = {0, 0, 0, 0, 0, 0, 0, 0};
  __builtin_amdgcn_tensor_load_to_lds(g0, g1, z4, z4, z8, 0);
#else
  __builtin_amdgcn_tensor_load_to_lds(g0, g1, z4, z4, 0);
#endif
}
#endif

// ---------------------------------------------------------------------------
// Helpers
// ---------------------------------------------------------------------------
__device__ __forceinline__ unsigned short f2bf(float f) {
  unsigned u = __float_as_uint(f);
  u += 0x7FFFu + ((u >> 16) & 1u);          // round-to-nearest-even
  return (unsigned short)(u >> 16);
}
__device__ __forceinline__ float bf2f(unsigned short b) {
  return __uint_as_float(((unsigned)b) << 16);
}
__device__ __forceinline__ float wave_sum(float v) {
#pragma unroll
  for (int o = 16; o > 0; o >>= 1) v += __shfl_xor(v, o, 32);
  return v;
}
__device__ __forceinline__ float wave_max(float v) {
#pragma unroll
  for (int o = 16; o > 0; o >>= 1) v = fmaxf(v, __shfl_xor(v, o, 32));
  return v;
}

// ---------------------------------------------------------------------------
// Elementwise / setup kernels
// ---------------------------------------------------------------------------
__global__ void k_embed(const int* __restrict__ ids,
                        const float* __restrict__ ew,
                        float* __restrict__ h) {
  int t  = blockIdx.x;
  int id = ids[t];
  const float* src = ew + (long long)id * kD;
  float* dst       = h  + (long long)t  * kD;
  for (int d = threadIdx.x; d < kD; d += 256) dst[d] = src[d];
}

__global__ void k_convert(const float* __restrict__ x,
                          unsigned short* __restrict__ y, long long n) {
  long long i = (long long)blockIdx.x * 256 + threadIdx.x;
  if (i < n) y[i] = f2bf(x[i]);
}

__global__ void k_gelu_bf16(const float* __restrict__ x,
                            unsigned short* __restrict__ y, long long n) {
  long long i = (long long)blockIdx.x * 256 + threadIdx.x;
  if (i < n) {
    float v = x[i];
    float u = 0.7978845608028654f * (v + 0.044715f * v * v * v);
    y[i] = f2bf(0.5f * v * (1.0f + tanhf(u)));
  }
}

__global__ void k_fill_f32(float* __restrict__ p, float v, long long n) {
  long long i = (long long)blockIdx.x * 256 + threadIdx.x;
  if (i < n) p[i] = v;
}
__global__ void k_fill_i32(int* __restrict__ p, int v, long long n) {
  long long i = (long long)blockIdx.x * 256 + threadIdx.x;
  if (i < n) p[i] = v;
}

// ---------------------------------------------------------------------------
// Router: logits -> softmax(9) -> top-2 mask -> masked gate per real expert
// one wave (32 lanes) per token
// ---------------------------------------------------------------------------
__global__ void k_router(const float* __restrict__ h,
                         const float* __restrict__ rw,   // (kNR, kD)
                         float* __restrict__ gate) {      // (kE, kT): prob or -inf
  int t    = blockIdx.x * 8 + (threadIdx.x >> 5);
  int lane = threadIdx.x & 31;
  if (t >= kT) return;
  float acc[kNR];
#pragma unroll
  for (int e = 0; e < kNR; ++e) acc[e] = 0.f;
  const float* hp = h + (long long)t * kD;
  for (int d = lane; d < kD; d += 32) {
    float x = hp[d];
#pragma unroll
    for (int e = 0; e < kNR; ++e) acc[e] += x * rw[e * kD + d];
  }
#pragma unroll
  for (int e = 0; e < kNR; ++e) acc[e] = wave_sum(acc[e]);
  if (lane == 0) {
    float m = -INFINITY;
#pragma unroll
    for (int e = 0; e < kNR; ++e) m = fmaxf(m, acc[e]);
    float p[kNR]; float s = 0.f;
#pragma unroll
    for (int e = 0; e < kNR; ++e) { p[e] = __expf(acc[e] - m); s += p[e]; }
    float inv = 1.f / s;
#pragma unroll
    for (int e = 0; e < kNR; ++e) p[e] *= inv;
    int i1 = 0;
    for (int e = 1; e < kNR; ++e) if (acc[e] > acc[i1]) i1 = e;
    int i2 = (i1 == 0) ? 1 : 0;
    for (int e = 0; e < kNR; ++e)
      if (e != i1 && acc[e] > acc[i2]) i2 = e;
#pragma unroll
    for (int e = 0; e < kE; ++e)
      gate[e * kT + t] = (e == i1 || e == i2) ? p[e] : -INFINITY;
  }
}

// ---------------------------------------------------------------------------
// Capacity selection by exact rank: token kept iff rank < CAP; slot = rank.
// Gate table (16KB) staged into LDS by the Tensor Data Mover when available.
// ---------------------------------------------------------------------------
__global__ void k_rank(const float* __restrict__ gate,
                       int* __restrict__ slot_tok,      // (kE, kCAP) preset -1
                       float* __restrict__ keptgate) {  // (kE, kT)
  int e = blockIdx.y;
  __shared__ float sg[kT];
  const float* gsrc = gate + (long long)e * kT;
#if defined(HAVE_TDM)
  if (threadIdx.x < 32) {                 // wave 0 drives the TDM descriptor
    tdm_load_1d(gsrc, sg, kT, 2 /*4-byte elements*/);
    __builtin_amdgcn_s_wait_tensorcnt(0);
  }
#else
  for (int i = threadIdx.x; i < kT; i += 256) sg[i] = gsrc[i];
#endif
  __syncthreads();
  int t  = blockIdx.x * 256 + threadIdx.x;
  float g = sg[t];
  float kg = 0.f;
  if (g != -INFINITY) {
    int rank = 0;
    for (int j = 0; j < kT; ++j) {
      float gj = sg[j];
      rank += (gj > g) || (gj == g && j < t);
    }
    if (rank < kCAP) { slot_tok[e * kCAP + rank] = t; kg = g; }
  }
  keptgate[e * kT + t] = kg;
}

// ---------------------------------------------------------------------------
// Gather dispatched tokens into bf16 expert buffers (zero for empty slots)
// ---------------------------------------------------------------------------
__global__ void k_gather(const float* __restrict__ h,
                         const int* __restrict__ slot_tok,
                         unsigned short* __restrict__ xb) {
  int row = blockIdx.x;                // e*CAP + c
  int t   = slot_tok[row];
  unsigned short* dst = xb + (long long)row * kD;
  if (t < 0) {
    for (int d = threadIdx.x; d < kD; d += 256) dst[d] = 0;
    return;
  }
  const float* src = h + (long long)t * kD;
  for (int d = threadIdx.x; d < kD; d += 256) dst[d] = f2bf(src[d]);
}

// ---------------------------------------------------------------------------
// Generic batched bf16 WMMA GEMM:  C(f32, MxN) = A(bf16, MxK) * B(bf16)
//   bT == 0 : B is K x N row-major (ldb)   |   bT == 1 : B is N x K (C=A*B^T)
// batch z: offset = (z>>zsh)*s?1 + (z & ((1<<zsh)-1))*s?2
// 256 threads = 8 waves (2x4), tile 128x128x32, wave tile 64x32 (4x2 WMMA),
// double-buffered LDS with async global->LDS staging where tiles are full.
// ---------------------------------------------------------------------------
__global__ __launch_bounds__(256) void k_gemm(
    int M, int N, int Kd,
    const unsigned short* __restrict__ A, int lda, long long sA1, long long sA2,
    const unsigned short* __restrict__ B, int ldb, long long sB1, long long sB2,
    float* __restrict__ C, int ldc, long long sC1, long long sC2,
    int zsh, int bT, float scale) {
  constexpr int BM = 128, BN = 128, BK = 32;
  __shared__ __align__(32) unsigned short As[2][BM * BK];
  __shared__ __align__(32) unsigned short Bs[2][BN * BK];

  int z = blockIdx.z;
  long long z1 = z >> zsh;
  long long z2 = z & ((1 << zsh) - 1);
  A += z1 * sA1 + z2 * sA2;
  B += z1 * sB1 + z2 * sB2;
  C += z1 * sC1 + z2 * sC2;

  int bm = blockIdx.y * BM, bn = blockIdx.x * BN;
  int tid = threadIdx.x, lane = tid & 31, w = tid >> 5;
  int wm = (w >> 2) * 64, wn = (w & 3) * 32;
  int lr = lane & 15, lh = lane >> 4;

  v8f acc[4][2];
#pragma unroll
  for (int mi = 0; mi < 4; ++mi)
#pragma unroll
    for (int ni = 0; ni < 2; ++ni)
#pragma unroll
      for (int j = 0; j < 8; ++j) acc[mi][ni][j] = 0.0f;

  int la_m = tid >> 1;             // 0..127 : tile row (A) / tile col (B)
  int la_k = (tid & 1) * 16;       // 0 or 16

  bool useAsyncA = false, useAsyncB = false;
#if defined(HAVE_ASYNC_COPY)
  useAsyncA = (bm + BM <= M);                 // full A tile (always true here)
  useAsyncB = (bT != 0) && (bn + BN <= N);    // full transposed-B tile
#endif
  const int acnt = (useAsyncA ? 2 : 0) + (useAsyncB ? 2 : 0);

  auto stageA = [&](int buf, int k0) {
    int gr = bm + la_m;
    unsigned short* dst = &As[buf][la_m * BK + la_k];
    const unsigned short* ap = A + (long long)gr * lda + k0 + la_k;
    if (useAsyncA) {
#if defined(HAVE_ASYNC_COPY)
      g2lds_b128(ap, dst);
      g2lds_b128(ap + 8, dst + 8);
#endif
    } else {
      unsigned short tmp[16];
      if (gr < M) {
        if (k0 + BK < Kd) __builtin_prefetch(ap + BK, 0, 1);
#pragma unroll
        for (int j = 0; j < 16; ++j) tmp[j] = ap[j];
      } else {
#pragma unroll
        for (int j = 0; j < 16; ++j) tmp[j] = 0;
      }
#pragma unroll
      for (int j = 0; j < 16; ++j) dst[j] = tmp[j];
    }
  };

  auto stageB = [&](int buf, int k0) {
    int gc = bn + la_m;
    unsigned short* dst = &Bs[buf][la_m * BK + la_k];
    if (useAsyncB) {
#if defined(HAVE_ASYNC_COPY)
      const unsigned short* bp = B + (long long)gc * ldb + k0 + la_k;
      g2lds_b128(bp, dst);
      g2lds_b128(bp + 8, dst + 8);
#endif
    } else {
      unsigned short tmp[16];
      if (gc < N) {
        if (bT) {
          const unsigned short* bp = B + (long long)gc * ldb + k0 + la_k;
#pragma unroll
          for (int j = 0; j < 16; ++j) tmp[j] = bp[j];
        } else {
#pragma unroll
          for (int j = 0; j < 16; ++j)
            tmp[j] = B[(long long)(k0 + la_k + j) * ldb + gc];
        }
      } else {
#pragma unroll
        for (int j = 0; j < 16; ++j) tmp[j] = 0;
      }
#pragma unroll
      for (int j = 0; j < 16; ++j) dst[j] = tmp[j];
    }
  };

  const int nk = Kd / BK;                // K is always a multiple of 32 here
  stageA(0, 0);
  stageB(0, 0);

  for (int it = 0; it < nk; ++it) {
    int cur = it & 1;
    if (it + 1 < nk) {
      stageA(1 - cur, (it + 1) * BK);    // async copies land in other buffer
      stageB(1 - cur, (it + 1) * BK);
      wait_async_n(acnt);                // tile `it` complete for this wave
    } else {
      wait_async_n(0);
    }
    __syncthreads();                     // all waves' tile `it` visible

    v16bf af[4], bfr[2];
#pragma unroll
    for (int mi = 0; mi < 4; ++mi) {
      int r = wm + mi * 16 + lr;
      af[mi] = *reinterpret_cast<const v16bf*>(&As[cur][r * BK + lh * 16]);
    }
#pragma unroll
    for (int ni = 0; ni < 2; ++ni) {
      int cix = wn + ni * 16 + lr;
      bfr[ni] = *reinterpret_cast<const v16bf*>(&Bs[cur][cix * BK + lh * 16]);
    }
#pragma unroll
    for (int mi = 0; mi < 4; ++mi)
#pragma unroll
      for (int ni = 0; ni < 2; ++ni)
        acc[mi][ni] = __builtin_amdgcn_wmma_f32_16x16x32_bf16(
            false, af[mi], false, bfr[ni], (short)0, acc[mi][ni], false, false);
    __syncthreads();                     // protect buffer reuse at it+2
  }

  // Store: C/D layout — VGPR i: M=i (lanes 0-15), M=i+8 (lanes 16-31); N=lane%16
#pragma unroll
  for (int mi = 0; mi < 4; ++mi) {
#pragma unroll
    for (int ni = 0; ni < 2; ++ni) {
      int col = bn + wn + ni * 16 + lr;
#pragma unroll
      for (int i = 0; i < 8; ++i) {
        int row = bm + wm + mi * 16 + lh * 8 + i;
        if (row < M && col < N)
          C[(long long)row * ldc + col] = acc[mi][ni][i] * scale;
      }
    }
  }
}

// ---------------------------------------------------------------------------
// RoPE on q,k (f32 in) -> bf16 out; angles computed analytically from token id
// ---------------------------------------------------------------------------
__global__ void k_rope(const float* __restrict__ qf,
                       const float* __restrict__ kf,
                       const int* __restrict__ slot_tok,
                       unsigned short* __restrict__ qb,
                       unsigned short* __restrict__ kb) {
  long long i = (long long)blockIdx.x * 256 + threadIdx.x;
  const long long n = 4LL * kCAP * kD;
  if (i >= n) return;
  int d  = (int)(i % kDH);
  long long r2 = (i / kDH) / kH;
  int c  = (int)(r2 % kCAP);
  int g  = (int)(r2 / kCAP);
  int t  = slot_tok[(2 * g) * kCAP + c];
  if (t < 0) { qb[i] = 0; kb[i] = 0; return; }
  const int half = kDH / 2;
  int j = d % half;
  float inv = __expf(-(2.0f * j / (float)kDH) * 9.210340371976184f); // 10000^-2j/dh
  float ang = (float)t * inv;
  float cs = __cosf(ang), sn = __sinf(ang);
  long long pidx = (d < half) ? (i + half) : (i - half);
  float q  = qf[i], kv = kf[i];
  float rq = (d < half) ? -qf[pidx] : qf[pidx];
  float rk = (d < half) ? -kf[pidx] : kf[pidx];
  qb[i] = f2bf(q * cs + rq * sn);
  kb[i] = f2bf(kv * cs + rk * sn);
}

// ---------------------------------------------------------------------------
// Row softmax (scale 1/sqrt(DH)) over CAP, bf16 output; one wave per row
// ---------------------------------------------------------------------------
__global__ void k_softmax(const float* __restrict__ s,
                          unsigned short* __restrict__ p) {
  int row  = blockIdx.x * 8 + (threadIdx.x >> 5);
  int lane = threadIdx.x & 31;
  const float scale = 0.125f;                 // 1/sqrt(64)
  const float* sp = s + (long long)row * kCAP;
  float m = -INFINITY;
  for (int j = lane; j < kCAP; j += 32) m = fmaxf(m, sp[j] * scale);
  m = wave_max(m);
  float sum = 0.f;
  for (int j = lane; j < kCAP; j += 32) sum += __expf(sp[j] * scale - m);
  sum = wave_sum(sum);
  float inv = 1.f / sum;
  unsigned short* pp = p + (long long)row * kCAP;
  for (int j = lane; j < kCAP; j += 32)
    pp[j] = f2bf(__expf(sp[j] * scale - m) * inv);
}

// ---------------------------------------------------------------------------
// Scatter expert outputs weighted by kept gates into combine buffer
// ---------------------------------------------------------------------------
__global__ void k_scatter(const float* __restrict__ eo,
                          const int* __restrict__ slot_tok,
                          const float* __restrict__ keptgate,
                          float* __restrict__ comb) {
  int row = blockIdx.x;                // e*CAP + c
  int e   = row / kCAP;
  int t   = slot_tok[row];
  if (t < 0) return;
  float wgt = keptgate[e * kT + t];
  const float* src = eo + (long long)row * kD;
  float* dst = comb + (long long)t * kD;
  for (int d = threadIdx.x; d < kD; d += 256)
    atomicAdd(&dst[d], wgt * src[d]);
}

__global__ void k_update(float* __restrict__ h,
                         const float* __restrict__ comb,
                         const float* __restrict__ keptgate) {
  int t = blockIdx.x;
  float rho = 0.f;
#pragma unroll
  for (int e = 0; e < kE; ++e) rho += keptgate[e * kT + t];
  float* hp = h + (long long)t * kD;
  const float* cp = comb + (long long)t * kD;
  for (int d = threadIdx.x; d < kD; d += 256)
    hp[d] = hp[d] * (1.f - rho) + cp[d];
}

__global__ void k_rmsnorm(const float* __restrict__ h,
                          const float* __restrict__ wgt,
                          float* __restrict__ out) {
  int t    = blockIdx.x * 8 + (threadIdx.x >> 5);
  int lane = threadIdx.x & 31;
  const float* hp = h + (long long)t * kD;
  float ss = 0.f;
  for (int d = lane; d < kD; d += 32) { float x = hp[d]; ss += x * x; }
  ss = wave_sum(ss);
  float r = rsqrtf(ss / (float)kD + 1e-6f);
  float* op = out + (long long)t * kD;
  for (int d = lane; d < kD; d += 32) op[d] = hp[d] * r * wgt[d];
}

// ---------------------------------------------------------------------------
// Host
// ---------------------------------------------------------------------------
extern "C" void kernel_launch(void* const* d_in, const int* in_sizes, int n_in,
                              void* d_out, int out_size, void* d_ws, size_t ws_size,
                              hipStream_t stream) {
  (void)in_sizes; (void)n_in; (void)out_size; (void)ws_size;

  const int*   ids      = (const int*)d_in[0];
  const float* embed_w  = (const float*)d_in[1];
  const float* router_w = (const float*)d_in[2];
  const float* wq       = (const float*)d_in[3];
  const float* wk       = (const float*)d_in[4];
  const float* wv       = (const float*)d_in[5];
  const float* wo       = (const float*)d_in[6];
  const float* w1       = (const float*)d_in[7];
  const float* w2       = (const float*)d_in[8];
  const float* ln_w     = (const float*)d_in[9];
  float* out = (float*)d_out;

  char* ws = (char*)d_ws;
  size_t off = 0;
  auto alloc = [&](size_t bytes) -> void* {
    void* p = ws + off;
    off = (off + bytes + 255) & ~(size_t)255;
    return p;
  };

  float*          h        = (float*)alloc((size_t)kT * kD * 4);
  float*          gate     = (float*)alloc((size_t)kE * kT * 4);
  float*          keptgate = (float*)alloc((size_t)kE * kT * 4);
  int*            slot     = (int*)alloc((size_t)kE * kCAP * 4);
  unsigned short* xinb     = (unsigned short*)alloc((size_t)kE * kCAP * kD * 2);
  unsigned short* wqb      = (unsigned short*)alloc(4ull * kD * kD * 2);
  unsigned short* wkb      = (unsigned short*)alloc(4ull * kD * kD * 2);
  unsigned short* wvb      = (unsigned short*)alloc(4ull * kD * kD * 2);
  unsigned short* wob      = (unsigned short*)alloc(4ull * kD * kD * 2);
  unsigned short* w1b      = (unsigned short*)alloc(4ull * kD * kMLP * 2);
  unsigned short* w2b      = (unsigned short*)alloc(4ull * kMLP * kD * 2);
  float*          qf       = (float*)alloc(4ull * kCAP * kD * 4);
  float*          kff      = (float*)alloc(4ull * kCAP * kD * 4);
  float*          vf       = (float*)alloc(4ull * kCAP * kD * 4);
  unsigned short* qb       = (unsigned short*)alloc(4ull * kCAP * kD * 2);
  unsigned short* kb       = (unsigned short*)alloc(4ull * kCAP * kD * 2);
  unsigned short* vb       = (unsigned short*)alloc(4ull * kCAP * kD * 2);
  float*          scores   = (float*)alloc(4ull * kH * kCAP * kCAP * 4);
  unsigned short* pb       = (unsigned short*)alloc(4ull * kH * kCAP * kCAP * 2);
  float*          aof      = (float*)alloc(4ull * kCAP * kD * 4);
  unsigned short* ab       = (unsigned short*)alloc(4ull * kCAP * kD * 2);
  float*          h1       = (float*)alloc(4ull * kCAP * kMLP * 4);
  unsigned short* h1b      = (unsigned short*)alloc(4ull * kCAP * kMLP * 2);
  float*          eo       = (float*)alloc((size_t)kE * kCAP * kD * 4);
  float*          comb     = (float*)alloc((size_t)kT * kD * 4);

  auto cdiv = [](long long a, long long b) { return (int)((a + b - 1) / b); };

  auto gemm = [&](int M, int N, int Kd,
                  const unsigned short* A, int lda, long long sA1, long long sA2,
                  const unsigned short* B, int ldb, long long sB1, long long sB2,
                  float* C, int ldc, long long sC1, long long sC2,
                  int zsh, int nz, int bT, float scale) {
    dim3 g(cdiv(N, 128), cdiv(M, 128), nz);
    k_gemm<<<g, 256, 0, stream>>>(M, N, Kd, A, lda, sA1, sA2,
                                  B, ldb, sB1, sB2, C, ldc, sC1, sC2,
                                  zsh, bT, scale);
  };

  // Convert all weights to bf16 once (resident in L2 afterwards)
  k_convert<<<cdiv(4ll * kD * kD, 256), 256, 0, stream>>>(wq, wqb, 4ll * kD * kD);
  k_convert<<<cdiv(4ll * kD * kD, 256), 256, 0, stream>>>(wk, wkb, 4ll * kD * kD);
  k_convert<<<cdiv(4ll * kD * kD, 256), 256, 0, stream>>>(wv, wvb, 4ll * kD * kD);
  k_convert<<<cdiv(4ll * kD * kD, 256), 256, 0, stream>>>(wo, wob, 4ll * kD * kD);
  k_convert<<<cdiv(4ll * kD * kMLP, 256), 256, 0, stream>>>(w1, w1b, 4ll * kD * kMLP);
  k_convert<<<cdiv(4ll * kMLP * kD, 256), 256, 0, stream>>>(w2, w2b, 4ll * kMLP * kD);

  k_embed<<<kT, 256, 0, stream>>>(ids, embed_w, h);

  for (int hop = 0; hop < kHOPS; ++hop) {
    // ---- route + dispatch ----
    k_router<<<kT / 8, 256, 0, stream>>>(h, router_w + (size_t)hop * kNR * kD, gate);
    k_fill_i32<<<cdiv(kE * kCAP, 256), 256, 0, stream>>>(slot, -1, (long long)kE * kCAP);
    k_rank<<<dim3(kT / 256, kE), 256, 0, stream>>>(gate, slot, keptgate);
    k_gather<<<kE * kCAP, 256, 0, stream>>>(h, slot, xinb);

    // ---- attention experts (groups g=0..3 -> experts 0,2,4,6) ----
    gemm(kCAP, kD, kD, xinb, kD, 2ll * kCAP * kD, 0,
         wqb, kD, (long long)kD * kD, 0,
         qf, kD, (long long)kCAP * kD, 0, 0, 4, 0, 1.f);
    gemm(kCAP, kD, kD, xinb, kD, 2ll * kCAP * kD, 0,
         wkb, kD, (long long)kD * kD, 0,
         kff, kD, (long long)kCAP * kD, 0, 0, 4, 0, 1.f);
    gemm(kCAP, kD, kD, xinb, kD, 2ll * kCAP * kD, 0,
         wvb, kD, (long long)kD * kD, 0,
         vf, kD, (long long)kCAP * kD, 0, 0, 4, 0, 1.f);

    k_rope<<<cdiv(4ll * kCAP * kD, 256), 256, 0, stream>>>(qf, kff, slot, qb, kb);
    k_convert<<<cdiv(4ll * kCAP * kD, 256), 256, 0, stream>>>(vf, vb, 4ll * kCAP * kD);

    // scores: z = g*16 + head (64 batches), S = Q * K^T (per head slice)
    gemm(kCAP, kCAP, kDH,
         qb, kD, (long long)kCAP * kD, (long long)kDH,
         kb, kD, (long long)kCAP * kD, (long long)kDH,
         scores, kCAP, (long long)kH * kCAP * kCAP, (long long)kCAP * kCAP,
         4, 64, 1, 1.f);

    k_softmax<<<(4 * kH * kCAP) / 8, 256, 0, stream>>>(scores, pb);

    // out = P * V (per head slice)
    gemm(kCAP, kDH, kCAP,
         pb, kCAP, (long long)kH * kCAP * kCAP, (long long)kCAP * kCAP,
         vb, kD, (long long)kCAP * kD, (long long)kDH,
         aof, kD, (long long)kCAP * kD, (long long)kDH,
         4, 64, 0, 1.f);

    k_convert<<<cdiv(4ll * kCAP * kD, 256), 256, 0, stream>>>(aof, ab, 4ll * kCAP * kD);

    // O projection -> expert_out at experts 0,2,4,6
    gemm(kCAP, kD, kD, ab, kD, (long long)kCAP * kD, 0,
         wob, kD, (long long)kD * kD, 0,
         eo, kD, 2ll * kCAP * kD, 0, 0, 4, 0, 1.f);

    // ---- FFN experts (groups g=0..3 -> experts 1,3,5,7) ----
    gemm(kCAP, kMLP, kD, xinb + (size_t)kCAP * kD, kD, 2ll * kCAP * kD, 0,
         w1b, kMLP, (long long)kD * kMLP, 0,
         h1, kMLP, (long long)kCAP * kMLP, 0, 0, 4, 0, 1.f);
    k_gelu_bf16<<<cdiv(4ll * kCAP * kMLP, 256), 256, 0, stream>>>(h1, h1b, 4ll * kCAP * kMLP);
    gemm(kCAP, kD, kMLP, h1b, kMLP, (long long)kCAP * kMLP, 0,
         w2b, kD, (long long)kMLP * kD, 0,
         eo + (size_t)kCAP * kD, kD, 2ll * kCAP * kD, 0, 0, 4, 0, 1.f);

    // ---- combine ----
    k_fill_f32<<<cdiv((long long)kT * kD, 256), 256, 0, stream>>>(comb, 0.f, (long long)kT * kD);
    k_scatter<<<kE * kCAP, 256, 0, stream>>>(eo, slot, keptgate, comb);
    k_update<<<kT, 256, 0, stream>>>(h, comb, keptgate);
  }

  k_rmsnorm<<<kT / 8, 256, 0, stream>>>(h, ln_w, out);
}
